// DGI_Hetero_28192165331398
// MI455X (gfx1250) — matrix operand order repaired and checked by reading, not compiled, not run on hardware
//
#include <hip/hip_runtime.h>
#include <hip/hip_bf16.h>

#define N_GENE 20000
#define N_CELL 50000
#define H_DIM  64

typedef __attribute__((ext_vector_type(16))) _Float16 v16h;
typedef __attribute__((ext_vector_type(8)))  float    v8f;

union FragH { v16h v; unsigned int u[8]; };

__device__ __forceinline__ int permIdx(int i, int n) {
    // deterministic bijection standing in for jax.random.permutation
    return (int)(((unsigned long long)(unsigned)i * 7919ull + 12345ull) % (unsigned long long)(unsigned)n);
}

__device__ __forceinline__ float log_sigmoid(float x) {
    float m = x < 0.f ? x : 0.f;
    return m - log1pf(expf(-fabsf(x)));
}

// ---------------- degree / scatter kernels ----------------

__global__ void dgi_degree(const int* __restrict__ dst, int E, float* __restrict__ cnt) {
    int e = blockIdx.x * blockDim.x + threadIdx.x;
    if (e < E) atomicAdd(&cnt[dst[e]], 1.0f);
}

__global__ void dgi_sage_scatter(const float* __restrict__ x, const int* __restrict__ src,
                                 const int* __restrict__ dst, int E,
                                 float* __restrict__ agg3, int permuted, int pN) {
    int e = blockIdx.x * blockDim.x + threadIdx.x;
    if (e >= E) return;
    int s = src[e];
    if (permuted) s = permIdx(s, pN);
    int d = dst[e];
    atomicAdd(&agg3[d * 3 + 0], x[s * 3 + 0]);
    atomicAdd(&agg3[d * 3 + 1], x[s * 3 + 1]);
    atomicAdd(&agg3[d * 3 + 2], x[s * 3 + 2]);
}

// ---------------- NNConv edge-message kernel (WMMA) ----------------
// Per edge: h = relu(ea*W1+b1) [32]; msg = outer(h,x)·W2perm + x·b2perm.
// A tile: 16 edges x 128 (cols 0..95 = h[k]*x[f] at k*3+f, 96..98 = x, rest 0)
// B (fixed per launch): 128x64 f16, staged in LDS in WMMA fragment order.
__global__ __launch_bounds__(128)
void dgi_nnconv_wmma(const float* __restrict__ xsrc, const int* __restrict__ src,
                     const int* __restrict__ dst, const float* __restrict__ ea,
                     const float* __restrict__ W1, const float* __restrict__ b1,
                     const float* __restrict__ W2, const float* __restrict__ b2,
                     float* __restrict__ aggH, int E, int permuted, int nSrc) {
    __shared__ alignas(32) _Float16 sBf[4 * 4 * 32 * 16];   // 16 KB, fragment-ordered B
    __shared__ alignas(32) _Float16 sA[4][16 * 128];        // 16 KB, per-wave A tiles
    __shared__ float sW1[32], sb1[32];
    __shared__ int sDst[4][16];

    const int tid  = threadIdx.x;
    const int wid  = tid >> 5;
    const int lane = tid & 31;

    // Stage W1/b1
    if (tid < 32) { sW1[tid] = W1[tid]; sb1[tid] = b1[tid]; }

    // Stage B in fragment order: element i of frag(kt,nt) on lane L is
    // B[kt*32 + (L>>4)*16 + i][nt*16 + (L&15)]
    for (int idx = tid; idx < 8192; idx += blockDim.x) {
        int i  = idx & 15;
        int ln = (idx >> 4) & 31;
        int nt = (idx >> 9) & 3;
        int kt = idx >> 11;
        int r  = kt * 32 + ((ln >> 4) ? 16 : 0) + i;   // K row in [0,128)
        int hc = nt * 16 + (ln & 15);                  // N col in [0,64)
        float val;
        if (r < 96) {
            int k = r / 3, f = r - k * 3;
            val = W2[k * 192 + f * 64 + hc];
        } else if (r < 99) {
            val = b2[(r - 96) * 64 + hc];
        } else {
            val = 0.f;
        }
        sBf[idx] = (_Float16)val;
    }
    __syncthreads();

    const int half = lane >> 4;
    const int mrow = lane & 15;
    _Float16* sAw = sA[wid];
    const unsigned int* sAu = (const unsigned int*)sAw;
    int* sDstW = sDst[wid];

    long nTiles  = ((long)E + 15) >> 4;
    long gw      = (long)blockIdx.x * (blockDim.x >> 5) + wid;
    long wstride = (long)gridDim.x * (blockDim.x >> 5);

    for (long t = gw; t < nTiles; t += wstride) {
        long e0 = t << 4;
        // ---- build A tile (lanes 0..15, one edge each) ----
        if (lane < 16) {
            long e = e0 + lane;
            _Float16* row = sAw + lane * 128;
            if (e < (long)E) {
                int s = src[e];
                if (permuted) s = permIdx(s, nSrc);
                float x0 = xsrc[s * 3 + 0], x1 = xsrc[s * 3 + 1], x2 = xsrc[s * 3 + 2];
                float a = ea[e];
#pragma unroll
                for (int k = 0; k < 32; ++k) {
                    float hk = a * sW1[k] + sb1[k];
                    hk = hk > 0.f ? hk : 0.f;
                    row[k * 3 + 0] = (_Float16)(hk * x0);
                    row[k * 3 + 1] = (_Float16)(hk * x1);
                    row[k * 3 + 2] = (_Float16)(hk * x2);
                }
                row[96] = (_Float16)x0; row[97] = (_Float16)x1; row[98] = (_Float16)x2;
#pragma unroll
                for (int c = 99; c < 128; ++c) row[c] = (_Float16)0.f;
                sDstW[lane] = dst[e];
            } else {
#pragma unroll
                for (int c = 0; c < 128; ++c) row[c] = (_Float16)0.f;
                sDstW[lane] = -1;
            }
        }
        // intra-wave LDS write->read ordering (same-wave DS ops are in-order;
        // the explicit wait also fences compiler reordering)
        asm volatile("s_wait_dscnt 0" ::: "memory");

        // ---- load A fragments (contiguous k = 16*half + i, matching B) ----
        FragH af[4];
#pragma unroll
        for (int kt = 0; kt < 4; ++kt) {
#pragma unroll
            for (int j = 0; j < 8; ++j) {
                int c = kt * 32 + half * 16 + j * 2;
                af[kt].u[j] = sAu[mrow * 64 + (c >> 1)];
            }
        }

        // ---- 4 N-tiles x 4 K-steps of WMMA, then scatter-mean (atomic add) ----
#pragma unroll
        for (int nt = 0; nt < 4; ++nt) {
            v8f acc = {};
#pragma unroll
            for (int kt = 0; kt < 4; ++kt) {
                v16h bf = *(const v16h*)(sBf + ((kt * 4 + nt) * 32 + lane) * 16);
                acc = __builtin_amdgcn_wmma_f32_16x16x32_f16(
                    false, af[kt].v, false, bf, (short)0, acc, false, false);
            }
            int n = nt * 16 + mrow;
#pragma unroll
            for (int r = 0; r < 8; ++r) {
                int m  = r + half * 8;           // C/D layout: VGPR r, lanes>=16 -> M+8
                int dn = sDstW[m];
                if (dn >= 0) atomicAdd(&aggH[dn * 64 + n], acc[r]);
            }
        }
    }
}

// ---------------- node combine: z = 0.5*(sage + nnconv) ----------------

__global__ void dgi_combine(const float* __restrict__ x,
                            const float* __restrict__ agg3, const float* __restrict__ cntS,
                            const float* __restrict__ aggH, const float* __restrict__ cntN,
                            const float* __restrict__ Wl, const float* __restrict__ bl,
                            const float* __restrict__ Wr,
                            const float* __restrict__ root, const float* __restrict__ bias,
                            float* __restrict__ z, int N, int permuted, int pN) {
    int idx = blockIdx.x * blockDim.x + threadIdx.x;
    if (idx >= N * 64) return;
    int node = idx >> 6, h = idx & 63;
    int xn = permuted ? permIdx(node, pN) : node;
    float x0 = x[xn * 3 + 0], x1 = x[xn * 3 + 1], x2 = x[xn * 3 + 2];

    float cs = cntS[node]; cs = cs > 1.f ? cs : 1.f;
    float a0 = agg3[node * 3 + 0] / cs;
    float a1 = agg3[node * 3 + 1] / cs;
    float a2 = agg3[node * 3 + 2] / cs;
    float sage = a0 * Wl[h] + a1 * Wl[64 + h] + a2 * Wl[128 + h] + bl[h]
               + x0 * Wr[h] + x1 * Wr[64 + h] + x2 * Wr[128 + h];

    float cn = cntN[node]; cn = cn > 1.f ? cn : 1.f;
    float nn = aggH[idx] / cn
             + x0 * root[h] + x1 * root[64 + h] + x2 * root[128 + h] + bias[h];

    z[idx] = 0.5f * (sage + nn);
}

// ---------------- readout / loss ----------------

__global__ void dgi_colsum(const float* __restrict__ z, int N, float* __restrict__ mean) {
    int tid = blockIdx.x * blockDim.x + threadIdx.x;
    int h = tid & 63;
    int r0 = tid >> 6;
    int stride = (gridDim.x * blockDim.x) >> 6;
    float s = 0.f;
    for (int r = r0; r < N; r += stride) s += z[r * 64 + h];
    atomicAdd(&mean[h], s);
}

__global__ void dgi_readout(const float* __restrict__ mean, float invN,
                            const float* __restrict__ W, const float* __restrict__ b,
                            float* __restrict__ summ) {
    int h = threadIdx.x;
    if (h >= 64) return;
    float s = b[h];
    for (int f = 0; f < 64; ++f) s += mean[f] * invN * W[f * 64 + h];
    summ[h] = s;
}

__global__ void dgi_score(const float* __restrict__ pos, const float* __restrict__ neg,
                          const float* __restrict__ summ, int N, float* __restrict__ acc) {
    __shared__ float sp[256], sn[256], ss[64];
    int tid = threadIdx.x;
    if (tid < 64) ss[tid] = summ[tid];
    __syncthreads();
    float lp = 0.f, ln = 0.f;
    for (int i = blockIdx.x * blockDim.x + tid; i < N; i += gridDim.x * blockDim.x) {
        float dp = 0.f, dn = 0.f;
#pragma unroll 8
        for (int h = 0; h < 64; ++h) {
            dp += pos[(long)i * 64 + h] * ss[h];
            dn += neg[(long)i * 64 + h] * ss[h];
        }
        lp += log_sigmoid(dp);
        ln += log_sigmoid(-dn);
    }
    sp[tid] = lp; sn[tid] = ln;
    __syncthreads();
    for (int s = 128; s > 0; s >>= 1) {
        if (tid < s) { sp[tid] += sp[tid + s]; sn[tid] += sn[tid + s]; }
        __syncthreads();
    }
    if (tid == 0) { atomicAdd(&acc[0], sp[0]); atomicAdd(&acc[1], sn[0]); }
}

__global__ void dgi_finalize(const float* __restrict__ acc, float* __restrict__ out) {
    if (threadIdx.x == 0 && blockIdx.x == 0)
        out[0] = -(acc[0] + acc[1]) / (float)N_GENE
               - (acc[2] + acc[3]) / (float)N_CELL;
}

// ---------------- host-side pipeline ----------------

extern "C" void kernel_launch(void* const* d_in, const int* in_sizes, int n_in,
                              void* d_out, int out_size, void* d_ws, size_t ws_size,
                              hipStream_t stream) {
    (void)in_sizes; (void)n_in; (void)out_size; (void)ws_size;

    const float* x_gene  = (const float*)d_in[0];
    const float* x_cell  = (const float*)d_in[1];
    const int*   gg_src  = (const int*)d_in[2];
    const int*   gg_dst  = (const int*)d_in[3];
    const int*   cc_src  = (const int*)d_in[4];
    const int*   cc_dst  = (const int*)d_in[5];
    const int*   cg_src  = (const int*)d_in[6];
    const int*   cg_dst  = (const int*)d_in[7];
    const int*   gc_src  = (const int*)d_in[8];
    const int*   gc_dst  = (const int*)d_in[9];
    const float* ea_cg   = (const float*)d_in[10];
    const float* ea_gc   = (const float*)d_in[11];
    const float* sage_gg_Wl = (const float*)d_in[12];
    const float* sage_gg_bl = (const float*)d_in[13];
    const float* sage_gg_Wr = (const float*)d_in[14];
    const float* sage_cc_Wl = (const float*)d_in[15];
    const float* sage_cc_bl = (const float*)d_in[16];
    const float* sage_cc_Wr = (const float*)d_in[17];
    const float* enn_W1  = (const float*)d_in[18];
    const float* enn_b1  = (const float*)d_in[19];
    const float* enn_W2  = (const float*)d_in[20];
    const float* enn_b2  = (const float*)d_in[21];
    const float* nn_cg_root = (const float*)d_in[22];
    const float* nn_cg_bias = (const float*)d_in[23];
    const float* nn_gc_root = (const float*)d_in[24];
    const float* nn_gc_bias = (const float*)d_in[25];
    const float* ro_gene_W  = (const float*)d_in[26];
    const float* ro_gene_b  = (const float*)d_in[27];
    const float* ro_cell_W  = (const float*)d_in[28];
    const float* ro_cell_b  = (const float*)d_in[29];

    const int E = in_sizes[2];   // 500000

    float* out   = (float*)d_out;
    float* pos_g = out + 1;
    float* pos_c = pos_g + (long)N_GENE * 64;

    float* ws     = (float*)d_ws;
    float* neg_g  = ws;                          // N_GENE*64
    float* neg_c  = neg_g + (long)N_GENE * 64;   // N_CELL*64
    float* agg3_g = neg_c + (long)N_CELL * 64;   // N_GENE*3
    float* agg3_c = agg3_g + (long)N_GENE * 3;   // N_CELL*3
    float* aggH_g = agg3_c + (long)N_CELL * 3;   // N_GENE*64
    float* aggH_c = aggH_g + (long)N_GENE * 64;  // N_CELL*64
    float* cnt_gg = aggH_c + (long)N_CELL * 64;  // N_GENE
    float* cnt_cc = cnt_gg + N_GENE;             // N_CELL
    float* cnt_cg = cnt_cc + N_CELL;             // N_GENE
    float* cnt_gc = cnt_cg + N_GENE;             // N_CELL
    float* mean_g = cnt_gc + N_CELL;             // 64
    float* mean_c = mean_g + 64;                 // 64
    float* summ_g = mean_c + 64;                 // 64
    float* summ_c = summ_g + 64;                 // 64
    float* acc    = summ_c + 64;                 // 4

    const int TB = 256;
    const int edgeBlocks = (E + TB - 1) / TB;

    // in-degree counts (independent of features, computed once per call)
    hipMemsetAsync(cnt_gg, 0, sizeof(float) * (N_GENE * 2 + N_CELL * 2), stream);
    hipMemsetAsync(mean_g, 0, sizeof(float) * (4 * 64 + 4), stream);
    dgi_degree<<<edgeBlocks, TB, 0, stream>>>(gg_dst, E, cnt_gg);
    dgi_degree<<<edgeBlocks, TB, 0, stream>>>(cc_dst, E, cnt_cc);
    dgi_degree<<<edgeBlocks, TB, 0, stream>>>(cg_dst, E, cnt_cg);
    dgi_degree<<<edgeBlocks, TB, 0, stream>>>(gc_dst, E, cnt_gc);

    for (int p = 0; p < 2; ++p) {   // p=0: positive, p=1: corrupted (permuted rows)
        hipMemsetAsync(agg3_g, 0,
            sizeof(float) * ((long)N_GENE * 3 + (long)N_CELL * 3 +
                             (long)N_GENE * 64 + (long)N_CELL * 64), stream);

        dgi_sage_scatter<<<edgeBlocks, TB, 0, stream>>>(x_gene, gg_src, gg_dst, E,
                                                        agg3_g, p, N_GENE);
        dgi_sage_scatter<<<edgeBlocks, TB, 0, stream>>>(x_cell, cc_src, cc_dst, E,
                                                        agg3_c, p, N_CELL);

        dgi_nnconv_wmma<<<1024, 128, 0, stream>>>(x_cell, cg_src, cg_dst, ea_cg,
                                                  enn_W1, enn_b1, enn_W2, enn_b2,
                                                  aggH_g, E, p, N_CELL);
        dgi_nnconv_wmma<<<1024, 128, 0, stream>>>(x_gene, gc_src, gc_dst, ea_gc,
                                                  enn_W1, enn_b1, enn_W2, enn_b2,
                                                  aggH_c, E, p, N_GENE);

        float* zg = p ? neg_g : pos_g;
        float* zc = p ? neg_c : pos_c;
        dgi_combine<<<(N_GENE * 64 + TB - 1) / TB, TB, 0, stream>>>(
            x_gene, agg3_g, cnt_gg, aggH_g, cnt_cg,
            sage_gg_Wl, sage_gg_bl, sage_gg_Wr, nn_cg_root, nn_cg_bias,
            zg, N_GENE, p, N_GENE);
        dgi_combine<<<(N_CELL * 64 + TB - 1) / TB, TB, 0, stream>>>(
            x_cell, agg3_c, cnt_cc, aggH_c, cnt_gc,
            sage_cc_Wl, sage_cc_bl, sage_cc_Wr, nn_gc_root, nn_gc_bias,
            zc, N_CELL, p, N_CELL);
    }

    // summaries from positive embeddings
    dgi_colsum<<<64, 256, 0, stream>>>(pos_g, N_GENE, mean_g);
    dgi_colsum<<<64, 256, 0, stream>>>(pos_c, N_CELL, mean_c);
    dgi_readout<<<1, 64, 0, stream>>>(mean_g, 1.0f / (float)N_GENE, ro_gene_W, ro_gene_b, summ_g);
    dgi_readout<<<1, 64, 0, stream>>>(mean_c, 1.0f / (float)N_CELL, ro_cell_W, ro_cell_b, summ_c);

    dgi_score<<<128, 256, 0, stream>>>(pos_g, neg_g, summ_g, N_GENE, acc);
    dgi_score<<<128, 256, 0, stream>>>(pos_c, neg_c, summ_c, N_CELL, acc + 2);
    dgi_finalize<<<1, 32, 0, stream>>>(acc, out);
}
